// Block_59433757442280
// MI455X (gfx1250) — 16-bit and 8-bit matrix operand order repaired and checked by reading, compiled, not run on hardware
//
#include <hip/hip_runtime.h>
#include <hip/hip_bf16.h>


typedef _Float16 half_t;
typedef _Float16 h8   __attribute__((ext_vector_type(8)));
typedef _Float16 v16h __attribute__((ext_vector_type(16)));
typedef float    v8f  __attribute__((ext_vector_type(8)));
typedef int      v16i __attribute__((ext_vector_type(16)));
typedef int      v4i  __attribute__((ext_vector_type(4)));

#define TLEN 4
#define BBAT 8
#define NTOK 1024
#define CCH  768
#define HDIM 3072
#define NHEAD 8
#define DHEAD 96
#define WIN   8

__device__ inline void async_b128(unsigned lds_byte_off, unsigned long long gaddr) {
    asm volatile("global_load_async_to_lds_b128 %0, %1, off"
                 :: "v"(lds_byte_off), "v"(gaddr) : "memory");
}
__device__ inline void wait_async0() { asm volatile("s_wait_asynccnt 0x0" ::: "memory"); }
__device__ inline void wait_async3() { asm volatile("s_wait_asynccnt 0x3" ::: "memory"); }
__device__ inline void wait_async6() { asm volatile("s_wait_asynccnt 0x6" ::: "memory"); }

__global__ void cvt_f32_f16(const float* __restrict__ src, half_t* __restrict__ dst, long n) {
    long i = (long)blockIdx.x * blockDim.x + threadIdx.x;
    if (i < n) dst[i] = (half_t)src[i];
}

__global__ void cvt_t_f16(const float* __restrict__ src, half_t* __restrict__ dst, int N, int K) {
    long i = (long)blockIdx.x * blockDim.x + threadIdx.x;
    if (i >= (long)N * K) return;
    int n = (int)(i / K), k = (int)(i % K);
    dst[(size_t)k * N + n] = (half_t)src[i];
}

__device__ inline unsigned char f32_to_e4m3(float x) {
    float ax = fabsf(x);
    unsigned s = (__float_as_uint(x) >> 31) << 7;
    if (ax < 0.001953125f) return (unsigned char)s;
    if (ax > 448.0f) ax = 448.0f;
    unsigned u = __float_as_uint(ax);
    u += 0x000FFFFF + ((u >> 20) & 1);
    int e = (int)((u >> 23) & 0xFF) - 127;
    unsigned m = (u >> 20) & 0x7;
    if (e < -6) return (unsigned char)s;
    if (e > 8 || (e == 8 && m == 7)) { e = 8; m = 6; }
    return (unsigned char)(s | ((unsigned)(e + 7) << 3) | m);
}

__global__ void cvt_t_fp8(const float* __restrict__ src, unsigned char* __restrict__ dst, int N, int K) {
    long i = (long)blockIdx.x * blockDim.x + threadIdx.x;
    if (i >= (long)N * K) return;
    int n = (int)(i / K), k = (int)(i % K);
    dst[(size_t)k * N + n] = f32_to_e4m3(src[i] * 64.0f);
}

__device__ inline v16h load_fragA_f16(const half_t* As_, int rowoff, int lane) {
    int r  = rowoff + (lane & 15);
    int kh = (lane >> 4) << 3;
    const half_t* p = As_ + r * 32 + kh;
    h8 lo = *(const h8*)(p);
    h8 hi = *(const h8*)(p + 16);
    return __builtin_shufflevector(lo, hi, 0,1,2,3,4,5,6,7,8,9,10,11,12,13,14,15);
}
__device__ inline v16h load_fragB_f16(const half_t* Bst, int n0, int lane) {
    const half_t* p = Bst + ((lane >> 4) << 3) * 72 + n0 + (lane & 15);
    v16h f;
#pragma unroll
    for (int s = 0; s < 8; ++s) { f[s] = p[s * 72]; f[s + 8] = p[(s + 16) * 72]; }
    return f;
}
__device__ inline v16i load_fragA_fp8(const unsigned char* As_, int rowoff, int lane) {
    int r   = rowoff + (lane & 15);
    int hi8 = (lane >> 4) << 3;
    const unsigned char* p = As_ + r * 128 + hi8;
    v16i f;
#pragma unroll
    for (int g = 0; g < 8; ++g) {
        int2 d = *(const int2*)(p + g * 16);
        f[2 * g]     = d.x;
        f[2 * g + 1] = d.y;
    }
    return f;
}
__device__ inline v16i load_fragB_fp8(const unsigned char* Bs_, int n0, int lane) {
    v16i f;
    const unsigned char* p = Bs_ + ((lane >> 4) << 3) * 80 + n0 + (lane & 15);
#pragma unroll
    for (int g = 0; g < 16; ++g) {
        const unsigned char* q = p + ((g >> 1) * 16 + (g & 1) * 4) * 80;
        f[g] = (int)((unsigned)q[0] | ((unsigned)q[80] << 8) | ((unsigned)q[160] << 16) | ((unsigned)q[240] << 24));
    }
    return f;
}

__global__ __launch_bounds__(256) void gemm_bn_f16(
    const half_t* __restrict__ A, const half_t* __restrict__ Wt,
    const float* __restrict__ bias, const float* __restrict__ scale,
    const float* __restrict__ shift, float* __restrict__ out,
    int M, int N, int K)
{
    __shared__ __align__(16) half_t As[2][128 * 32];
    __shared__ __align__(16) half_t Bst[2][32 * 72];

    int tid = threadIdx.x, lane = tid & 31, wave = tid >> 5;
    int m0 = blockIdx.y * 128, n0 = blockIdx.x * 64;
    int wm = wave & 3, wn = wave >> 2;

    int arow = tid >> 1, acol = (tid & 1) << 4;
    int brow = tid >> 3, bcol = (tid & 7) << 3;

    unsigned ldsA[2] = { (unsigned)(size_t)(&As[0][arow * 32 + acol]),
                         (unsigned)(size_t)(&As[1][arow * 32 + acol]) };
    unsigned ldsB[2] = { (unsigned)(size_t)(&Bst[0][brow * 72 + bcol]),
                         (unsigned)(size_t)(&Bst[1][brow * 72 + bcol]) };
    const half_t* gA = A + (size_t)(m0 + arow) * K + acol;
    const half_t* gB = Wt + (size_t)brow * N + n0 + bcol;

    v8f acc00 = {}; v8f acc01 = {}; v8f acc10 = {}; v8f acc11 = {};

    auto issue = [&](int kt, int b) {
        long k0 = (long)kt * 32;
        async_b128(ldsA[b],      (unsigned long long)(size_t)(gA + k0));
        async_b128(ldsA[b] + 16, (unsigned long long)(size_t)(gA + k0 + 8));
        async_b128(ldsB[b],      (unsigned long long)(size_t)(gB + k0 * N));
    };
    auto compute = [&](const half_t* Asb, const half_t* Bsb) {
        v16h a0 = load_fragA_f16(Asb, wm * 32,      lane);
        v16h a1 = load_fragA_f16(Asb, wm * 32 + 16, lane);
        v16h b0 = load_fragB_f16(Bsb, wn * 32,      lane);
        v16h b1 = load_fragB_f16(Bsb, wn * 32 + 16, lane);
        acc00 = __builtin_amdgcn_wmma_f32_16x16x32_f16(false, a0, false, b0, (short)0, acc00, false, false);
        acc01 = __builtin_amdgcn_wmma_f32_16x16x32_f16(false, a0, false, b1, (short)0, acc01, false, false);
        acc10 = __builtin_amdgcn_wmma_f32_16x16x32_f16(false, a1, false, b0, (short)0, acc10, false, false);
        acc11 = __builtin_amdgcn_wmma_f32_16x16x32_f16(false, a1, false, b1, (short)0, acc11, false, false);
    };

    int nt = K >> 5;
    issue(0, 0);
    for (int it = 0; it < nt - 1; ++it) {
        issue(it + 1, (it + 1) & 1);
        wait_async3();
        __syncthreads();
        compute(As[it & 1], Bst[it & 1]);
        __syncthreads();
    }
    wait_async0();
    __syncthreads();
    compute(As[(nt - 1) & 1], Bst[(nt - 1) & 1]);

    int nloc0 = n0 + wn * 32 + (lane & 15);
    int nloc1 = nloc0 + 16;
    float bb0 = bias[nloc0], ss0 = scale[nloc0], tt0 = shift[nloc0];
    float bb1 = bias[nloc1], ss1 = scale[nloc1], tt1 = shift[nloc1];
    int mb0 = m0 + wm * 32 + ((lane >> 4) << 3);
    int mb1 = mb0 + 16;
#pragma unroll
    for (int i = 0; i < 8; ++i) {
        size_t r0 = (size_t)(mb0 + i) * N;
        size_t r1 = (size_t)(mb1 + i) * N;
        out[r0 + nloc0] = (acc00[i] + bb0) * ss0 + tt0;
        out[r0 + nloc1] = (acc01[i] + bb1) * ss1 + tt1;
        out[r1 + nloc0] = (acc10[i] + bb0) * ss0 + tt0;
        out[r1 + nloc1] = (acc11[i] + bb1) * ss1 + tt1;
    }
}

__global__ __launch_bounds__(256) void gemm_bn_fp8(
    const unsigned char* __restrict__ A, const unsigned char* __restrict__ Wt,
    const float* __restrict__ bias, const float* __restrict__ scale,
    const float* __restrict__ shift, float* __restrict__ out,
    int M, int N, int K)
{
    __shared__ __align__(16) unsigned char As[2][128 * 128];
    __shared__ __align__(16) unsigned char Bs[2][128 * 80];

    int tid = threadIdx.x, lane = tid & 31, wave = tid >> 5;
    int m0 = blockIdx.y * 128, n0 = blockIdx.x * 64;
    int wm = wave & 3, wn = wave >> 2;

    int arow = tid >> 1, acol = (tid & 1) << 6;
    int brow = tid >> 1, bcol = (tid & 1) << 5;

    unsigned ldsA[2] = { (unsigned)(size_t)(&As[0][arow * 128 + acol]),
                         (unsigned)(size_t)(&As[1][arow * 128 + acol]) };
    unsigned ldsB[2] = { (unsigned)(size_t)(&Bs[0][brow * 80 + bcol]),
                         (unsigned)(size_t)(&Bs[1][brow * 80 + bcol]) };
    const unsigned char* gA = A + (size_t)(m0 + arow) * K + acol;
    const unsigned char* gB = Wt + (size_t)brow * N + n0 + bcol;

    v8f acc00 = {}; v8f acc01 = {}; v8f acc10 = {}; v8f acc11 = {};

    auto issue = [&](int kt, int b) {
        long k0 = (long)kt * 128;
        async_b128(ldsA[b],      (unsigned long long)(size_t)(gA + k0));
        async_b128(ldsA[b] + 16, (unsigned long long)(size_t)(gA + k0 + 16));
        async_b128(ldsA[b] + 32, (unsigned long long)(size_t)(gA + k0 + 32));
        async_b128(ldsA[b] + 48, (unsigned long long)(size_t)(gA + k0 + 48));
        async_b128(ldsB[b],      (unsigned long long)(size_t)(gB + k0 * N));
        async_b128(ldsB[b] + 16, (unsigned long long)(size_t)(gB + k0 * N + 16));
    };
    auto compute = [&](const unsigned char* Asb, const unsigned char* Bsb) {
        v16i a0 = load_fragA_fp8(Asb, wm * 32,      lane);
        v16i a1 = load_fragA_fp8(Asb, wm * 32 + 16, lane);
        v16i b0 = load_fragB_fp8(Bsb, wn * 32,      lane);
        v16i b1 = load_fragB_fp8(Bsb, wn * 32 + 16, lane);
        acc00 = __builtin_amdgcn_wmma_f32_16x16x128_fp8_fp8(a0, b0, (short)0, acc00, false, false);
        acc01 = __builtin_amdgcn_wmma_f32_16x16x128_fp8_fp8(a0, b1, (short)0, acc01, false, false);
        acc10 = __builtin_amdgcn_wmma_f32_16x16x128_fp8_fp8(a1, b0, (short)0, acc10, false, false);
        acc11 = __builtin_amdgcn_wmma_f32_16x16x128_fp8_fp8(a1, b1, (short)0, acc11, false, false);
    };

    int nt = K >> 7;
    issue(0, 0);
    for (int it = 0; it < nt - 1; ++it) {
        issue(it + 1, (it + 1) & 1);
        wait_async6();
        __syncthreads();
        compute(As[it & 1], Bs[it & 1]);
        __syncthreads();
    }
    wait_async0();
    __syncthreads();
    compute(As[(nt - 1) & 1], Bs[(nt - 1) & 1]);

    const float invs = 0.015625f;
    int nloc0 = n0 + wn * 32 + (lane & 15);
    int nloc1 = nloc0 + 16;
    float bb0 = bias[nloc0], ss0 = scale[nloc0], tt0 = shift[nloc0];
    float bb1 = bias[nloc1], ss1 = scale[nloc1], tt1 = shift[nloc1];
    int mb0 = m0 + wm * 32 + ((lane >> 4) << 3);
    int mb1 = mb0 + 16;
#pragma unroll
    for (int i = 0; i < 8; ++i) {
        size_t r0 = (size_t)(mb0 + i) * N;
        size_t r1 = (size_t)(mb1 + i) * N;
        out[r0 + nloc0] = (acc00[i] * invs + bb0) * ss0 + tt0;
        out[r0 + nloc1] = (acc01[i] * invs + bb1) * ss1 + tt1;
        out[r1 + nloc0] = (acc10[i] * invs + bb0) * ss0 + tt0;
        out[r1 + nloc1] = (acc11[i] * invs + bb1) * ss1 + tt1;
    }
}

__global__ void lif_f16_kernel(const float* __restrict__ pre, half_t* __restrict__ spk,
                               long plane, float vth)
{
    long i = (long)blockIdx.x * blockDim.x + threadIdx.x;
    if (i >= plane) return;
    float v = 0.0f;
#pragma unroll
    for (int t = 0; t < TLEN; ++t) {
        float u = pre[(long)t * plane + i];
        v = 0.5f * (v + u);
        float s = (v >= vth) ? 1.0f : 0.0f;
        spk[(long)t * plane + i] = (half_t)s;
        v *= (1.0f - s);
    }
}

__global__ void lif_fp8_kernel(const float* __restrict__ pre, unsigned char* __restrict__ spk,
                               long plane, float vth)
{
    long i = (long)blockIdx.x * blockDim.x + threadIdx.x;
    if (i >= plane) return;
    float v = 0.0f;
#pragma unroll
    for (int t = 0; t < TLEN; ++t) {
        float u = pre[(long)t * plane + i];
        v = 0.5f * (v + u);
        bool s = (v >= vth);
        spk[(long)t * plane + i] = s ? (unsigned char)0x38 : (unsigned char)0x00;
        v = s ? 0.0f : v;
    }
}

__global__ void attn_kernel(const half_t* __restrict__ q, const half_t* __restrict__ k,
                            const half_t* __restrict__ v, float* __restrict__ o)
{
    int gid = blockIdx.x * blockDim.x + threadIdx.x;
    int n  = gid & (NTOK - 1);
    int h  = (gid >> 10) & (NHEAD - 1);
    int tb = gid >> 13;
    const float scl = 0.1020620726159658f;

    size_t rowbase = ((size_t)tb * NTOK + n) * CCH + (size_t)h * DHEAD;
    half_t qr[DHEAD];
#pragma unroll
    for (int d = 0; d < DHEAD; ++d) qr[d] = q[rowbase + d];

    int iw = n >> 3;
    float sim[16];
    float mx = -3.0e38f;
#pragma unroll
    for (int j = 0; j < 16; ++j) {
        if (iw == 0 && j < WIN) { sim[j] = -3.0e38f; continue; }
        int nk = (iw - 1) * WIN + j;
        size_t kb = ((size_t)tb * NTOK + nk) * CCH + (size_t)h * DHEAD;
        float acc = 0.0f;
#pragma unroll
        for (int d = 0; d < DHEAD; ++d) acc += (float)qr[d] * (float)k[kb + d];
        sim[j] = acc * scl;
        mx = fmaxf(mx, sim[j]);
    }
    float denom = 0.0f;
#pragma unroll
    for (int j = 0; j < 16; ++j) {
        float e = (sim[j] <= -1.0e38f) ? 0.0f : __expf(sim[j] - mx);
        sim[j] = e;
        denom += e;
    }
    float inv = 1.0f / denom;

    float outd[DHEAD];
#pragma unroll
    for (int d = 0; d < DHEAD; ++d) outd[d] = 0.0f;
#pragma unroll
    for (int j = 0; j < 16; ++j) {
        if (iw == 0 && j < WIN) continue;
        float aj = sim[j] * inv;
        int nk = (iw - 1) * WIN + j;
        size_t vb = ((size_t)tb * NTOK + nk) * CCH + (size_t)h * DHEAD;
#pragma unroll
        for (int d = 0; d < DHEAD; ++d) outd[d] += aj * (float)v[vb + d];
    }
#pragma unroll
    for (int d = 0; d < DHEAD; ++d) o[rowbase + d] = outd[d];
}

__global__ void residual_kernel(const float* __restrict__ x, const unsigned char* __restrict__ spk,
                                float* __restrict__ x1f, half_t* __restrict__ x1h, long n)
{
    long i = (long)blockIdx.x * blockDim.x + threadIdx.x;
    if (i >= n) return;
    float v = x[i] + (spk[i] ? 1.0f : 0.0f);
    x1f[i] = v;
    x1h[i] = (half_t)v;
}

__global__ void final_kernel(const float* __restrict__ x1f, const unsigned char* __restrict__ spk,
                             float* __restrict__ out, long n)
{
    long i = (long)blockIdx.x * blockDim.x + threadIdx.x;
    if (i >= n) return;
    out[i] = x1f[i] + (spk[i] ? 1.0f : 0.0f);
}

extern "C" void kernel_launch(void* const* d_in, const int* in_sizes, int n_in,
                              void* d_out, int out_size, void* d_ws, size_t ws_size,
                              hipStream_t stream)
{
    (void)in_sizes; (void)n_in; (void)out_size; (void)ws_size;
    const float* x   = (const float*)d_in[0];
    const float* qw  = (const float*)d_in[1];
    const float* qb  = (const float*)d_in[2];
    const float* qs  = (const float*)d_in[3];
    const float* qt  = (const float*)d_in[4];
    const float* kw  = (const float*)d_in[5];
    const float* kb  = (const float*)d_in[6];
    const float* ks  = (const float*)d_in[7];
    const float* kt  = (const float*)d_in[8];
    const float* vw  = (const float*)d_in[9];
    const float* vb  = (const float*)d_in[10];
    const float* vs  = (const float*)d_in[11];
    const float* vt  = (const float*)d_in[12];
    const float* pw  = (const float*)d_in[13];
    const float* pb  = (const float*)d_in[14];
    const float* ps  = (const float*)d_in[15];
    const float* pt  = (const float*)d_in[16];
    const float* f1w = (const float*)d_in[17];
    const float* f1b = (const float*)d_in[18];
    const float* f1s = (const float*)d_in[19];
    const float* f1t = (const float*)d_in[20];
    const float* f2w = (const float*)d_in[21];
    const float* f2b = (const float*)d_in[22];
    const float* f2s = (const float*)d_in[23];
    const float* f2t = (const float*)d_in[24];

    const long Mrows = (long)TLEN * BBAT * NTOK;
    const long Cc = CCH, Hd = HDIM;
    const long E  = Mrows * Cc;
    const long E2 = Mrows * Hd;
    const long planeC = (long)BBAT * NTOK * Cc;
    const long planeH = (long)BBAT * NTOK * Hd;

    half_t* Wqt  = (half_t*)d_ws;
    half_t* Wkt  = Wqt  + Cc * Cc;
    half_t* Wvt  = Wkt  + Cc * Cc;
    half_t* Wf1t = Wvt  + Cc * Cc;
    unsigned char* Wpt  = (unsigned char*)(Wf1t + Cc * Hd);
    unsigned char* Wf2t = Wpt + Cc * Cc;
    half_t* Xh   = (half_t*)(Wf2t + Hd * Cc);
    half_t* Qs   = Xh + E;
    half_t* Ks   = Qs + E;
    half_t* Vs   = Ks + E;
    unsigned char* oSpk = (unsigned char*)(Vs + E);
    unsigned char* pSpk = oSpk + E;
    unsigned char* mSpk = pSpk + E;
    float*  P    = (float*)(mSpk + E);
    float*  O    = P + E2;
    unsigned char* hSpk = (unsigned char*)Qs;

    dim3 blk(256);
    auto nb = [](long n) { return (unsigned)((n + 255) / 256); };

    cvt_f32_f16<<<nb(E), blk, 0, stream>>>(x, Xh, E);
    cvt_t_f16<<<nb(Cc * Cc), blk, 0, stream>>>(qw, Wqt, (int)Cc, (int)Cc);
    cvt_t_f16<<<nb(Cc * Cc), blk, 0, stream>>>(kw, Wkt, (int)Cc, (int)Cc);
    cvt_t_f16<<<nb(Cc * Cc), blk, 0, stream>>>(vw, Wvt, (int)Cc, (int)Cc);
    cvt_t_f16<<<nb(Hd * Cc), blk, 0, stream>>>(f1w, Wf1t, (int)Hd, (int)Cc);
    cvt_t_fp8<<<nb(Cc * Cc), blk, 0, stream>>>(pw, Wpt, (int)Cc, (int)Cc);
    cvt_t_fp8<<<nb(Cc * Hd), blk, 0, stream>>>(f2w, Wf2t, (int)Cc, (int)Hd);

    dim3 gC((unsigned)(Cc / 64), (unsigned)(Mrows / 128));
    dim3 gH((unsigned)(Hd / 64), (unsigned)(Mrows / 128));

    gemm_bn_f16<<<gC, blk, 0, stream>>>(Xh, Wqt, qb, qs, qt, P, (int)Mrows, (int)Cc, (int)Cc);
    lif_f16_kernel<<<nb(planeC), blk, 0, stream>>>(P, Qs, planeC, 1.0f);
    gemm_bn_f16<<<gC, blk, 0, stream>>>(Xh, Wkt, kb, ks, kt, P, (int)Mrows, (int)Cc, (int)Cc);
    lif_f16_kernel<<<nb(planeC), blk, 0, stream>>>(P, Ks, planeC, 1.0f);
    gemm_bn_f16<<<gC, blk, 0, stream>>>(Xh, Wvt, vb, vs, vt, P, (int)Mrows, (int)Cc, (int)Cc);
    lif_f16_kernel<<<nb(planeC), blk, 0, stream>>>(P, Vs, planeC, 1.0f);

    attn_kernel<<<(unsigned)((long)TLEN * BBAT * NHEAD * NTOK / 256), blk, 0, stream>>>(Qs, Ks, Vs, O);
    lif_fp8_kernel<<<nb(planeC), blk, 0, stream>>>(O, oSpk, planeC, 0.5f);

    gemm_bn_fp8<<<gC, blk, 0, stream>>>(oSpk, Wpt, pb, ps, pt, P, (int)Mrows, (int)Cc, (int)Cc);
    lif_fp8_kernel<<<nb(planeC), blk, 0, stream>>>(P, pSpk, planeC, 1.0f);

    residual_kernel<<<nb(E), blk, 0, stream>>>(x, pSpk, O, Xh, E);

    gemm_bn_f16<<<gH, blk, 0, stream>>>(Xh, Wf1t, f1b, f1s, f1t, P, (int)Mrows, (int)Hd, (int)Cc);
    lif_fp8_kernel<<<nb(planeH), blk, 0, stream>>>(P, hSpk, planeH, 1.0f);

    gemm_bn_fp8<<<gC, blk, 0, stream>>>(hSpk, Wf2t, f2b, f2s, f2t, P, (int)Mrows, (int)Cc, (int)Hd);
    lif_fp8_kernel<<<nb(planeC), blk, 0, stream>>>(P, mSpk, planeC, 1.0f);

    final_kernel<<<nb(E), blk, 0, stream>>>(O, mSpk, (float*)d_out, E);
}
